// RMCBlock_63359357551026
// MI455X (gfx1250) — matrix operand order repaired and checked by reading, not compiled, not run on hardware
//
#include <hip/hip_runtime.h>
#include <hip/hip_bf16.h>
#include <cstdint>
#include <cstddef>

// ---------------- model dims (fixed by the reference) ----------------
#define DIMN    2048
#define HIDDEN  2048
#define LOWRK   256
#define FFDIM   8192
#define NH      16
#define NKV     4
#define HD      128
#define NREP    4
#define BB      2
#define SSQ     2048
#define ROWS    (BB*SSQ)   // 4096

typedef __bf16 bf16_t;
typedef __attribute__((ext_vector_type(16))) __bf16 v16bf;
typedef __attribute__((ext_vector_type(8)))  __bf16 v8bf;
typedef __attribute__((ext_vector_type(8)))  float  v8f;

__device__ __forceinline__ bf16_t f2bf(float f) { return (bf16_t)f; }

// D = A(16x32 bf16) * B(32x16 bf16) + C(16x16 f32), wave32 striped
__device__ __forceinline__ v8f wmma_bf16(v16bf a, v16bf b, v8f c) {
  return __builtin_amdgcn_wmma_f32_16x16x32_bf16(
      /*neg_a=*/false, a, /*neg_b=*/false, b,
      /*c_mod=*/(short)0, c, /*reuse_a=*/false, /*reuse_b=*/false);
}

// A fragment (16x32, MxK) from row-major [M,K]: lane l<16 -> row l, K {0..7,16..23};
// lane>=16 -> row l-16, K {8..15,24..31}. Two contiguous 16B loads per lane.
__device__ __forceinline__ v16bf load_fragA(const bf16_t* __restrict__ p, int ld) {
  const int lane = threadIdx.x & 31;
  const bf16_t* base = p + (size_t)(lane & 15) * ld + ((lane >> 4) << 3);
  v8bf lo = *(const v8bf*)(base);
  v8bf hi = *(const v8bf*)(base + 16);
  v16bf r;
#pragma unroll
  for (int i = 0; i < 8; ++i) { r[i] = lo[i]; r[i + 8] = hi[i]; }
  return r;
}

// B fragment (32x16, KxN) where B(k,n) = W[n,k], W row-major [N,K]:
// lane n holds col n, K {0..7,16..23} (lane<16) / {8..15,24..31} (lane>=16) -> two 16B loads per lane.
__device__ __forceinline__ v16bf load_fragB(const bf16_t* __restrict__ p, int ld) {
  const int lane = threadIdx.x & 31;
  const bf16_t* base = p + (size_t)(lane & 15) * ld + ((lane >> 4) << 3);
  v8bf lo = *(const v8bf*)(base);
  v8bf hi = *(const v8bf*)(base + 16);
  v16bf r;
#pragma unroll
  for (int i = 0; i < 8; ++i) { r[i] = lo[i]; r[i + 8] = hi[i]; }
  return r;
}

__device__ __forceinline__ float act_gelu_ent(float h) {
  float g   = 0.5f * h * (1.0f + erff(h * 0.70710678118654752f));   // exact gelu
  float p   = 1.0f / (1.0f + __expf(-h));
  float ent = -(p * __logf(p + 1e-9f) + (1.0f - p) * __logf(1.0f - p + 1e-9f));
  return g - 0.05f * ent;
}

// ---------------- generic WMMA GEMM:  C[M,N] = A[M,K] @ W[N,K]^T (+bias)(+resid)(+act)
// Block = 256 threads = 8 waves; block tile 128x256; wave tile 64x64 (4x4 accum frags).
// Per 32-K step: 16 WMMAs vs 8 fragment loads -> 2x math density of a 64x32 tile.
// EPI: 0 = f32 out, 1 = bf16 out, 2 = f32 out + residual, 3 = bf16 out + gelu-entropy act
template <int EPI>
__global__ __launch_bounds__(256) void gemm_wmma_kernel(
    const bf16_t* __restrict__ A, const bf16_t* __restrict__ W,
    const float* __restrict__ bias, const float* resid,
    void* outp, int M, int N, int K) {
  const int lane   = threadIdx.x & 31;
  const int w      = threadIdx.x >> 5;
  const int m_base = blockIdx.y * 128 + (w >> 2) * 64;
  const int n_base = blockIdx.x * 256 + (w & 3) * 64;

  v8f acc[4][4];
#pragma unroll
  for (int mt = 0; mt < 4; ++mt)
#pragma unroll
    for (int nt = 0; nt < 4; ++nt)
#pragma unroll
      for (int i = 0; i < 8; ++i) acc[mt][nt][i] = 0.0f;

  const bf16_t* Arow = A + (size_t)(m_base + (lane & 15)) * K;        // per-lane A row
  const bf16_t* Wrow = W + (size_t)(n_base + (lane & 15)) * K;        // per-lane W row

  for (int k0 = 0; k0 < K; k0 += 32) {
    // prefetch 4 K-steps ahead (speculative; OOB translation drops silently)
    __builtin_prefetch(Arow + k0 + 128, 0, 1);
    __builtin_prefetch(Wrow + k0 + 128, 0, 1);
    v16bf bfr[4];
#pragma unroll
    for (int nt = 0; nt < 4; ++nt)
      bfr[nt] = load_fragB(W + (size_t)(n_base + nt * 16) * K + k0, K);
#pragma unroll
    for (int mt = 0; mt < 4; ++mt) {
      v16bf af = load_fragA(A + (size_t)(m_base + mt * 16) * K + k0, K);
#pragma unroll
      for (int nt = 0; nt < 4; ++nt)
        acc[mt][nt] = wmma_bf16(af, bfr[nt], acc[mt][nt]);
    }
  }

  const int rb = (lane >> 4) * 8;   // row base within 16x16 C frag
  const int cl = lane & 15;         // column within frag
#pragma unroll
  for (int mt = 0; mt < 4; ++mt) {
#pragma unroll
    for (int nt = 0; nt < 4; ++nt) {
      const int c = n_base + nt * 16 + cl;
      const float bv = bias ? bias[c] : 0.0f;
#pragma unroll
      for (int i = 0; i < 8; ++i) {
        const size_t r = (size_t)(m_base + mt * 16 + rb + i);
        const size_t o = r * (size_t)N + c;
        float v = acc[mt][nt][i] + bv;
        if (EPI == 0)      ((float*)outp)[o]  = v;
        else if (EPI == 1) ((bf16_t*)outp)[o] = f2bf(v);
        else if (EPI == 2) ((float*)outp)[o]  = resid[o] + v;
        else               ((bf16_t*)outp)[o] = f2bf(act_gelu_ent(v));
      }
    }
  }
}

// ---------------- flash attention (causal, GQA), bf16 WMMA, f32 online softmax
// grid = (S/128, B*NH); 8 waves/block, each wave owns 16 query rows independently.
__global__ __launch_bounds__(256) void attn_flash_kernel(
    const bf16_t* __restrict__ Q,   // [B,S,NH,HD]
    const bf16_t* __restrict__ Km,  // [B,S,NKV,HD]
    const bf16_t* __restrict__ Vt,  // [B,NKV,HD,S]
    bf16_t* __restrict__ O) {       // [B,S,NH,HD]
  __shared__ alignas(16) bf16_t lds_p[8][16][40];  // per-wave P staging, padded pitch

  const int lane = threadIdx.x & 31;
  const int w    = threadIdx.x >> 5;
  const int bh   = blockIdx.y;
  const int b    = bh / NH, h = bh % NH;
  const int kvh  = h / NREP;
  const int m0   = blockIdx.x * 128 + w * 16;
  const float scale = 0.088388347648318447f;  // 1/sqrt(128)

  const bf16_t* Qbase = Q + ((size_t)(b * SSQ + m0)) * (NH * HD) + h * HD;
  v16bf qf[4];
#pragma unroll
  for (int dc = 0; dc < 4; ++dc) qf[dc] = load_fragA(Qbase + dc * 32, NH * HD);

  v8f acc[8];
#pragma unroll
  for (int nd = 0; nd < 8; ++nd)
#pragma unroll
    for (int i = 0; i < 8; ++i) acc[nd][i] = 0.0f;
  float m_i[8], l_i[8];
#pragma unroll
  for (int i = 0; i < 8; ++i) { m_i[i] = -3.0e38f; l_i[i] = 0.0f; }

  const int rb = (lane >> 4) * 8;
  const int cl = lane & 15;

  for (int n0 = 0; n0 < m0 + 16; n0 += 32) {   // causal tile range (overshoot masked)
    v8f s0, s1;
#pragma unroll
    for (int i = 0; i < 8; ++i) { s0[i] = 0.0f; s1[i] = 0.0f; }
    const bf16_t* Kbase = Km + ((size_t)(b * SSQ + n0)) * (NKV * HD) + kvh * HD;
#pragma unroll
    for (int dc = 0; dc < 4; ++dc) {
      v16bf k0f = load_fragB(Kbase + dc * 32, NKV * HD);
      v16bf k1f = load_fragB(Kbase + (size_t)16 * (NKV * HD) + dc * 32, NKV * HD);
      s0 = wmma_bf16(qf[dc], k0f, s0);
      s1 = wmma_bf16(qf[dc], k1f, s1);
    }
    float corr[8];
#pragma unroll
    for (int i = 0; i < 8; ++i) {
      const int grow = m0 + rb + i;
      float sv0 = s0[i] * scale + ((n0 + cl      <= grow) ? 0.0f : -1.0e9f);
      float sv1 = s1[i] * scale + ((n0 + 16 + cl <= grow) ? 0.0f : -1.0e9f);
      float tm = fmaxf(sv0, sv1);
#pragma unroll
      for (int msk = 1; msk < 16; msk <<= 1) tm = fmaxf(tm, __shfl_xor(tm, msk, 32));
      float nm = fmaxf(m_i[i], tm);
      float cr = __expf(m_i[i] - nm);
      float p0 = __expf(sv0 - nm);
      float p1 = __expf(sv1 - nm);
      float ps = p0 + p1;
#pragma unroll
      for (int msk = 1; msk < 16; msk <<= 1) ps += __shfl_xor(ps, msk, 32);
      l_i[i] = l_i[i] * cr + ps;
      m_i[i] = nm;
      corr[i] = cr;
      lds_p[w][rb + i][cl]      = f2bf(p0);   // C-layout -> LDS (cross-lane transpose)
      lds_p[w][rb + i][16 + cl] = f2bf(p1);
    }
#pragma unroll
    for (int nd = 0; nd < 8; ++nd)
#pragma unroll
      for (int i = 0; i < 8; ++i) acc[nd][i] *= corr[i];

    // P as A-fragment (16x32) from LDS; same-wave DS ops are in order.
    const bf16_t* pb = &lds_p[w][cl][rb];
    v16bf pf;
#pragma unroll
    for (int i = 0; i < 8; ++i) { pf[i] = pb[i]; pf[8 + i] = pb[16 + i]; }

    const bf16_t* Vb = Vt + ((size_t)(b * NKV + kvh)) * HD * (size_t)SSQ + n0;
#pragma unroll
    for (int nd = 0; nd < 8; ++nd) {
      v16bf vf = load_fragB(Vb + (size_t)(nd * 16) * SSQ, SSQ);
      acc[nd] = wmma_bf16(pf, vf, acc[nd]);
    }
  }

  bf16_t* Ob = O + ((size_t)(b * SSQ + m0)) * (NH * HD) + h * HD;
#pragma unroll
  for (int nd = 0; nd < 8; ++nd)
#pragma unroll
    for (int i = 0; i < 8; ++i)
      Ob[(size_t)(rb + i) * (NH * HD) + nd * 16 + cl] = f2bf(acc[nd][i] / l_i[i]);
}

// ---------------- small helper kernels ----------------
__global__ void cast_bf16_kernel(const float* __restrict__ in, bf16_t* __restrict__ out, int n) {
  int i = blockIdx.x * blockDim.x + threadIdx.x;
  if (i < n) out[i] = f2bf(in[i]);
}

__global__ __launch_bounds__(256) void rmsnorm_kernel(
    const float* __restrict__ xin, const float* __restrict__ w, bf16_t* __restrict__ out) {
  __shared__ float red[256];
  const size_t row = blockIdx.x;
  const float* xr = xin + row * DIMN;
  float ss = 0.0f;
  for (int i = threadIdx.x; i < DIMN; i += 256) { float v = xr[i]; ss += v * v; }
  red[threadIdx.x] = ss;
  __syncthreads();
  for (int st = 128; st > 0; st >>= 1) {
    if ((int)threadIdx.x < st) red[threadIdx.x] += red[threadIdx.x + st];
    __syncthreads();
  }
  const float rms = rsqrtf(red[0] / (float)DIMN + 1e-6f);
  bf16_t* o = out + row * DIMN;
  for (int i = threadIdx.x; i < DIMN; i += 256) o[i] = f2bf(xr[i] * rms * w[i]);
}

__global__ void rope_kernel(const float* __restrict__ xin, const float* __restrict__ c,
                            const float* __restrict__ s, bf16_t* __restrict__ out, int nh) {
  int idx = blockIdx.x * blockDim.x + threadIdx.x;  // < B*S*nh*64
  int d2  = idx & 63;
  int t   = idx >> 6;                 // (b*S + s)*nh + h
  int srw = (t / nh) & (SSQ - 1);
  size_t base = (size_t)t * HD;
  float cv = c[srw * 64 + d2], sv = s[srw * 64 + d2];
  float xe = xin[base + 2 * d2], xo = xin[base + 2 * d2 + 1];
  out[base + 2 * d2]     = f2bf(xe * cv - xo * sv);
  out[base + 2 * d2 + 1] = f2bf(xe * sv + xo * cv);
}

__global__ void vtranspose_kernel(const float* __restrict__ v, bf16_t* __restrict__ vt) {
  int idx = blockIdx.x * blockDim.x + threadIdx.x;  // < B*S*NKV*HD
  int d  = idx & (HD - 1);
  int t  = idx >> 7;
  int kv = t & (NKV - 1);
  int bs = t >> 2;
  int s  = bs & (SSQ - 1);
  int b  = bs >> 11;
  vt[((size_t)(b * NKV + kv) * HD + d) * SSQ + s] = f2bf(v[idx]);
}

__global__ void ssm_scan_kernel(const float* __restrict__ inp, const float* __restrict__ g,
                                const float* __restrict__ be, bf16_t* __restrict__ st) {
  int idx = blockIdx.x * blockDim.x + threadIdx.x;  // < B*HIDDEN
  int ch = idx & (HIDDEN - 1);
  int b  = idx >> 11;
  float gg = g[ch], bb = be[ch];
  float s = 0.0f;
  const float* ip = inp + (size_t)b * SSQ * HIDDEN + ch;
  bf16_t* op = st + (size_t)b * SSQ * HIDDEN + ch;
  for (int t = 0; t < SSQ; ++t) {
    // latency-bound serial chain: pull future timesteps toward the WGP early
    __builtin_prefetch(ip + (size_t)(t + 32) * HIDDEN, 0, 1);
    s = tanhf(ip[(size_t)t * HIDDEN] + s * gg + bb);
    op[(size_t)t * HIDDEN] = f2bf(s);
  }
}

// ---------------- launcher ----------------
extern "C" void kernel_launch(void* const* d_in, const int* in_sizes, int n_in,
                              void* d_out, int out_size, void* d_ws, size_t ws_size,
                              hipStream_t stream) {
  const float* x       = (const float*)d_in[0];
  const float* cosp    = (const float*)d_in[1];
  const float* sinp    = (const float*)d_in[2];
  // d_in[3] = mask: causal mask computed analytically in-kernel (identical values)
  const float* attn_nw = (const float*)d_in[4];
  const float* wq      = (const float*)d_in[5];
  const float* wk      = (const float*)d_in[6];
  const float* wv      = (const float*)d_in[7];
  const float* wo      = (const float*)d_in[8];
  const float* ssm_nw  = (const float*)d_in[9];
  const float* ssm_wi  = (const float*)d_in[10];
  const float* ssm_bi  = (const float*)d_in[11];
  const float* ssm_g   = (const float*)d_in[12];
  const float* ssm_be  = (const float*)d_in[13];
  const float* ssm_wo  = (const float*)d_in[14];
  const float* ssm_bo  = (const float*)d_in[15];
  const float* ffn_nw  = (const float*)d_in[16];
  const float* fc1_u   = (const float*)d_in[17];
  const float* fc1_v   = (const float*)d_in[18];
  const float* fc1_b   = (const float*)d_in[19];
  const float* fc2_u   = (const float*)d_in[20];
  const float* fc2_v   = (const float*)d_in[21];
  const float* fc2_b   = (const float*)d_in[22];

  char* ws = (char*)d_ws;
  size_t off = 0;
  auto alloc = [&](size_t bytes) -> char* {
    char* r = ws + off;
    off += (bytes + 255) & ~(size_t)255;
    return r;
  };

  // bf16 weights (cast every launch; deterministic, cheap)
  bf16_t* wq_bf  = (bf16_t*)alloc((size_t)DIMN * DIMN * 2);
  bf16_t* wk_bf  = (bf16_t*)alloc((size_t)512 * DIMN * 2);
  bf16_t* wv_bf  = (bf16_t*)alloc((size_t)512 * DIMN * 2);
  bf16_t* wo_bf  = (bf16_t*)alloc((size_t)DIMN * DIMN * 2);
  bf16_t* wi_bf  = (bf16_t*)alloc((size_t)HIDDEN * DIMN * 2);
  bf16_t* wo2_bf = (bf16_t*)alloc((size_t)DIMN * HIDDEN * 2);
  bf16_t* f1u_bf = (bf16_t*)alloc((size_t)LOWRK * DIMN * 2);
  bf16_t* f1v_bf = (bf16_t*)alloc((size_t)FFDIM * LOWRK * 2);
  bf16_t* f2u_bf = (bf16_t*)alloc((size_t)LOWRK * FFDIM * 2);
  bf16_t* f2v_bf = (bf16_t*)alloc((size_t)DIMN * LOWRK * 2);

  bf16_t* xn_bf = (bf16_t*)alloc((size_t)ROWS * DIMN * 2);  // 16MB, reused per stage
  // 64MB region with stage-local overlays
  char* big = alloc((size_t)ROWS * FFDIM * 2);
  float*  q_f32   = (float*)big;                                              // 32MB
  float*  k_f32   = (float*)(big + (size_t)ROWS * DIMN * 4);                  // 8MB
  float*  v_f32   = (float*)(big + (size_t)ROWS * DIMN * 4 + (size_t)ROWS * 512 * 4);
  bf16_t* qr_bf   = (bf16_t*)(big + (size_t)ROWS * DIMN * 4 + 2 * (size_t)ROWS * 512 * 4);
  float*  inp_f32 = (float*)big;      // SSM stage: q/k/v dead
  bf16_t* st_bf   = qr_bf;            // SSM stage: qr dead
  bf16_t* h_bf    = (bf16_t*)big;     // FFN stage: whole region dead

  bf16_t* kr_bf  = (bf16_t*)alloc((size_t)ROWS * 512 * 2);
  bf16_t* vt_bf  = (bf16_t*)alloc((size_t)ROWS * 512 * 2);
  bf16_t* ao_bf  = (bf16_t*)alloc((size_t)ROWS * DIMN * 2);
  float*  x1_f32 = (float*)alloc((size_t)ROWS * DIMN * 4);  // running residual
  bf16_t* t1_bf  = (bf16_t*)alloc((size_t)ROWS * LOWRK * 2);
  bf16_t* t2_bf  = kr_bf;  // FFN stage: kr dead

  const dim3 blk(256);
  auto cgrid = [](size_t n) { return dim3((unsigned)((n + 255) / 256)); };
  auto ggrid = [](int M, int N) { return dim3(N / 256, M / 128); };  // 128x256 block tiles

  // weight casts
  cast_bf16_kernel<<<cgrid((size_t)DIMN * DIMN), blk, 0, stream>>>(wq, wq_bf, DIMN * DIMN);
  cast_bf16_kernel<<<cgrid((size_t)512 * DIMN), blk, 0, stream>>>(wk, wk_bf, 512 * DIMN);
  cast_bf16_kernel<<<cgrid((size_t)512 * DIMN), blk, 0, stream>>>(wv, wv_bf, 512 * DIMN);
  cast_bf16_kernel<<<cgrid((size_t)DIMN * DIMN), blk, 0, stream>>>(wo, wo_bf, DIMN * DIMN);
  cast_bf16_kernel<<<cgrid((size_t)HIDDEN * DIMN), blk, 0, stream>>>(ssm_wi, wi_bf, HIDDEN * DIMN);
  cast_bf16_kernel<<<cgrid((size_t)DIMN * HIDDEN), blk, 0, stream>>>(ssm_wo, wo2_bf, DIMN * HIDDEN);
  cast_bf16_kernel<<<cgrid((size_t)LOWRK * DIMN), blk, 0, stream>>>(fc1_u, f1u_bf, LOWRK * DIMN);
  cast_bf16_kernel<<<cgrid((size_t)FFDIM * LOWRK), blk, 0, stream>>>(fc1_v, f1v_bf, FFDIM * LOWRK);
  cast_bf16_kernel<<<cgrid((size_t)LOWRK * FFDIM), blk, 0, stream>>>(fc2_u, f2u_bf, LOWRK * FFDIM);
  cast_bf16_kernel<<<cgrid((size_t)DIMN * LOWRK), blk, 0, stream>>>(fc2_v, f2v_bf, DIMN * LOWRK);

  // ---- attention ----
  rmsnorm_kernel<<<ROWS, blk, 0, stream>>>(x, attn_nw, xn_bf);
  gemm_wmma_kernel<0><<<ggrid(ROWS, DIMN), blk, 0, stream>>>(xn_bf, wq_bf, nullptr, nullptr, q_f32, ROWS, DIMN, DIMN);
  gemm_wmma_kernel<0><<<ggrid(ROWS, 512), blk, 0, stream>>>(xn_bf, wk_bf, nullptr, nullptr, k_f32, ROWS, 512, DIMN);
  gemm_wmma_kernel<0><<<ggrid(ROWS, 512), blk, 0, stream>>>(xn_bf, wv_bf, nullptr, nullptr, v_f32, ROWS, 512, DIMN);
  rope_kernel<<<cgrid((size_t)ROWS * NH * 64), blk, 0, stream>>>(q_f32, cosp, sinp, qr_bf, NH);
  rope_kernel<<<cgrid((size_t)ROWS * NKV * 64), blk, 0, stream>>>(k_f32, cosp, sinp, kr_bf, NKV);
  vtranspose_kernel<<<cgrid((size_t)ROWS * NKV * HD), blk, 0, stream>>>(v_f32, vt_bf);
  attn_flash_kernel<<<dim3(SSQ / 128, BB * NH), blk, 0, stream>>>(qr_bf, kr_bf, vt_bf, ao_bf);
  gemm_wmma_kernel<2><<<ggrid(ROWS, DIMN), blk, 0, stream>>>(ao_bf, wo_bf, nullptr, x, x1_f32, ROWS, DIMN, DIMN);

  // ---- SSM ----
  rmsnorm_kernel<<<ROWS, blk, 0, stream>>>(x1_f32, ssm_nw, xn_bf);
  gemm_wmma_kernel<0><<<ggrid(ROWS, HIDDEN), blk, 0, stream>>>(xn_bf, wi_bf, ssm_bi, nullptr, inp_f32, ROWS, HIDDEN, DIMN);
  ssm_scan_kernel<<<cgrid((size_t)BB * HIDDEN), blk, 0, stream>>>(inp_f32, ssm_g, ssm_be, st_bf);
  gemm_wmma_kernel<2><<<ggrid(ROWS, DIMN), blk, 0, stream>>>(st_bf, wo2_bf, ssm_bo, x1_f32, x1_f32, ROWS, DIMN, HIDDEN);

  // ---- FFN ----
  rmsnorm_kernel<<<ROWS, blk, 0, stream>>>(x1_f32, ffn_nw, xn_bf);
  gemm_wmma_kernel<1><<<ggrid(ROWS, LOWRK), blk, 0, stream>>>(xn_bf, f1u_bf, nullptr, nullptr, t1_bf, ROWS, LOWRK, DIMN);
  gemm_wmma_kernel<3><<<ggrid(ROWS, FFDIM), blk, 0, stream>>>(t1_bf, f1v_bf, fc1_b, nullptr, h_bf, ROWS, FFDIM, LOWRK);
  gemm_wmma_kernel<1><<<ggrid(ROWS, LOWRK), blk, 0, stream>>>(h_bf, f2u_bf, nullptr, nullptr, t2_bf, ROWS, LOWRK, FFDIM);
  gemm_wmma_kernel<2><<<ggrid(ROWS, DIMN), blk, 0, stream>>>(t2_bf, f2v_bf, fc2_b, x1_f32, (float*)d_out, ROWS, DIMN, LOWRK);
}
